// acLSTM_50818053046810
// MI455X (gfx1250) — matrix operand order repaired and checked by reading, not compiled, not run on hardware
//
#include <hip/hip_runtime.h>
#include <hip/hip_bf16.h>
#include <math.h>

typedef __attribute__((ext_vector_type(16))) __bf16 v16bf;
typedef __attribute__((ext_vector_type(8)))  float  v8f;
typedef __attribute__((ext_vector_type(4)))  unsigned int v4u;

#define DEV __device__ __forceinline__

// ---------------- problem constants ----------------
constexpr int HID   = 1024;
constexpr int FRAME = 171;
constexpr int BATCH = 32;
constexpr int T0    = 20;
constexpr int KX1   = 192;                 // frame padded to mult of 32
constexpr int KT1   = KX1/32 + HID/32;     // 38 k-tiles for layer1 ([x|h]) (even)
constexpr int KT2   = 2*HID/32;            // 64 k-tiles for layers 2,3     (even)
constexpr int KTD   = HID/32;              // 32 k-tiles for decoder        (even)
constexpr int NJT   = HID/16;              // 64 hidden-unit tiles
constexpr int NTD   = 11;                  // ceil(176/16) decoder col tiles
constexpr int DECPAD= 176;

constexpr int NBLOCKS  = 16;
constexpr int NTHREADS = 128;              // 4 wave32 per block -> 64 waves

// ---------------- workspace layout (bytes) ----------------
constexpr size_t PB1_E = (size_t)NJT*KT1*4*32*16;   // packed bf16 elements
constexpr size_t PB2_E = (size_t)NJT*KT2*4*32*16;
constexpr size_t PBD_E = (size_t)NTD*KTD*32*16;
constexpr size_t HSZB  = (size_t)BATCH*HID*2;       // one h buffer (bf16 bytes)

constexpr size_t alup(size_t x){ return (x + 255) & ~(size_t)255; }
constexpr size_t OFF_PB1  = 0;
constexpr size_t OFF_PB2  = alup(OFF_PB1 + PB1_E*2);
constexpr size_t OFF_PB3  = alup(OFF_PB2 + PB2_E*2);
constexpr size_t OFF_PBD  = alup(OFF_PB3 + PB2_E*2);
constexpr size_t OFF_B1   = alup(OFF_PBD + PBD_E*2);
constexpr size_t OFF_B2   = alup(OFF_B1 + 4096*4);
constexpr size_t OFF_B3   = alup(OFF_B2 + 4096*4);
constexpr size_t OFF_BD   = alup(OFF_B3 + 4096*4);
constexpr size_t OFF_X1   = alup(OFF_BD + DECPAD*4);            // 32 x 192 bf16
constexpr size_t OFF_H1   = alup(OFF_X1 + (size_t)BATCH*KX1*2); // 2 bufs
constexpr size_t OFF_A2H1 = alup(OFF_H1 + 2*HSZB);
constexpr size_t OFF_H2   = alup(OFF_A2H1 + HSZB);              // 2 bufs
constexpr size_t OFF_A3H2 = alup(OFF_H2 + 2*HSZB);
constexpr size_t OFF_H3   = alup(OFF_A3H2 + HSZB);              // 2 bufs
constexpr size_t OFF_ADEC = alup(OFF_H3 + 2*HSZB);
constexpr size_t OFF_INITX= alup(OFF_ADEC + HSZB);              // 20 x 32 x 192 bf16
constexpr size_t OFF_C    = alup(OFF_INITX + (size_t)T0*BATCH*KX1*2); // 3 x 32 x 1024 f32
constexpr size_t OFF_SYNC = alup(OFF_C + (size_t)3*BATCH*HID*4);
constexpr size_t WS_NEED  = OFF_SYNC + 256;

// ---------------- global-address-space access helpers ----------------
DEV v4u gload4(const unsigned short* p) {
  return *(const __attribute__((address_space(1))) v4u*)(const v4u*)p;
}
DEV float gloadf(const float* p) {
  return *(const __attribute__((address_space(1))) float*)p;
}
DEV unsigned short gloadh(const unsigned short* p) {
  return *(const __attribute__((address_space(1))) unsigned short*)p;
}
DEV void gstoref(float* p, float v) {
  *(__attribute__((address_space(1))) float*)p = v;
}
DEV void gstoreh(unsigned short* p, unsigned short v) {
  *(__attribute__((address_space(1))) unsigned short*)p = v;
}

// ---------------- misc helpers ----------------
DEV unsigned short f32_to_bf16(float f) {
  union { float f; unsigned u; } x; x.f = f;
  unsigned u = x.u;
  u += 0x7FFFu + ((u >> 16) & 1u);        // round-to-nearest-even
  return (unsigned short)(u >> 16);
}
DEV float sigm(float x) { return 1.0f / (1.0f + __expf(-x)); }

union Frag { v16bf v; v4u q[2]; };
struct FragSet { Frag a0, a1, b0, b1, b2, b3; };   // statically-named -> SROA to regs
struct DFragSet { Frag a0, a1, b; };

// device-wide sense barrier (persistent kernel; NBLOCKS small -> all resident)
DEV void grid_sync(int* cnt, int* phs, int target) {
  __threadfence();
  __syncthreads();
  if (threadIdx.x == 0) {
    if (__hip_atomic_fetch_add(cnt, 1, __ATOMIC_ACQ_REL, __HIP_MEMORY_SCOPE_AGENT) == NBLOCKS - 1) {
      __hip_atomic_store(cnt, 0, __ATOMIC_RELAXED, __HIP_MEMORY_SCOPE_AGENT);
      __hip_atomic_fetch_add(phs, 1, __ATOMIC_ACQ_REL, __HIP_MEMORY_SCOPE_AGENT);
    } else {
      while (__hip_atomic_load(phs, __ATOMIC_ACQUIRE, __HIP_MEMORY_SCOPE_AGENT) < target)
        __builtin_amdgcn_s_sleep(2);
    }
  }
  __syncthreads();
  __threadfence();
}

// ---------------- prep kernels ----------------
struct Ptrs {
  const float *iseq, *bih1, *bhh1, *bih2, *bhh2, *bih3, *bhh3, *bdec;
};

__global__ void misc_init(Ptrs p, char* __restrict__ ws) {
  float* b1 = (float*)(ws + OFF_B1);
  float* b2 = (float*)(ws + OFF_B2);
  float* b3 = (float*)(ws + OFF_B3);
  float* bd = (float*)(ws + OFF_BD);
  unsigned short* X1    = (unsigned short*)(ws + OFF_X1);
  unsigned short* initx = (unsigned short*)(ws + OFF_INITX);
  unsigned int*   zr    = (unsigned int*)(ws + OFF_H1);
  float*          cst   = (float*)(ws + OFF_C);
  int*            sync  = (int*)(ws + OFF_SYNC);

  size_t tid = (size_t)blockIdx.x * blockDim.x + threadIdx.x;
  size_t str = (size_t)gridDim.x * blockDim.x;

  for (size_t i = tid; i < 4096; i += str) {
    gstoref(b1 + i, gloadf(p.bih1 + i) + gloadf(p.bhh1 + i));
    gstoref(b2 + i, gloadf(p.bih2 + i) + gloadf(p.bhh2 + i));
    gstoref(b3 + i, gloadf(p.bih3 + i) + gloadf(p.bhh3 + i));
  }
  for (size_t i = tid; i < DECPAD; i += str)
    gstoref(bd + i, (i < FRAME) ? gloadf(p.bdec + i) : 0.0f);
  // initial_seq -> padded bf16 [t][m][192]
  for (size_t i = tid; i < (size_t)T0 * BATCH * KX1; i += str) {
    int f = (int)(i % KX1); int m = (int)((i / KX1) % BATCH); int t = (int)(i / ((size_t)KX1 * BATCH));
    gstoreh(initx + i, (f < FRAME) ? f32_to_bf16(gloadf(p.iseq + ((size_t)m * T0 + t) * FRAME + f))
                                   : (unsigned short)0);
  }
  // X1 = x(0)
  for (size_t i = tid; i < (size_t)BATCH * KX1; i += str) {
    int f = (int)(i % KX1); int m = (int)(i / KX1);
    gstoreh(X1 + i, (f < FRAME) ? f32_to_bf16(gloadf(p.iseq + ((size_t)m * T0 + 0) * FRAME + f))
                                : (unsigned short)0);
  }
  // zero all h staging buffers (H1x2, A2H1, H2x2, A3H2, H3x2, ADEC)
  for (size_t i = tid; i < (OFF_INITX - OFF_H1) / 4; i += str)
    *(__attribute__((address_space(1))) unsigned int*)(zr + i) = 0u;
  // zero c state
  for (size_t i = tid; i < (size_t)3 * BATCH * HID; i += str) gstoref(cst + i, 0.0f);
  // zero sync
  for (size_t i = tid; i < 64; i += str)
    *(__attribute__((address_space(1))) int*)(sync + i) = 0;
}

// Pack [W_ih | W_hh]^T into WMMA B-fragment order (bf16).
// element index: (((jt*KT + kt)*4 + gate)*32 + lane)*16 + i
// B layout (32x16, 16-bit): K = kt*32 + (i>>3)*16 + (lane>>4)*8 + (i&7) ; N = lane&15
__global__ void pack_gates(const float* __restrict__ Wih, int sih, int Kxpad, int Kxreal,
                           const float* __restrict__ Whh, int KT,
                           unsigned short* __restrict__ dst, size_t total) {
  size_t tid = (size_t)blockIdx.x * blockDim.x + threadIdx.x;
  size_t str = (size_t)gridDim.x * blockDim.x;
  for (size_t f = tid; f < total; f += str) {
    int i    = (int)(f & 15);
    int lane = (int)((f >> 4) & 31);
    int gate = (int)((f >> 9) & 3);
    size_t r = f >> 11;
    int kt = (int)(r % KT);
    int jt = (int)(r / KT);
    int n  = gate * HID + jt * 16 + (lane & 15);
    int k  = kt * 32 + ((i >> 3) << 4) + ((lane >> 4) & 1) * 8 + (i & 7);
    float v;
    if (k < Kxpad) v = (k < Kxreal) ? gloadf(Wih + (size_t)n * sih + k) : 0.0f;
    else           v = gloadf(Whh + (size_t)n * HID + (k - Kxpad));
    gstoreh(dst + f, f32_to_bf16(v));
  }
}

__global__ void pack_dec(const float* __restrict__ Wdec, unsigned short* __restrict__ dst) {
  size_t tid = (size_t)blockIdx.x * blockDim.x + threadIdx.x;
  size_t str = (size_t)gridDim.x * blockDim.x;
  for (size_t f = tid; f < PBD_E; f += str) {
    int i    = (int)(f & 15);
    int lane = (int)((f >> 4) & 31);
    size_t r = f >> 9;
    int kt = (int)(r % KTD);
    int jt = (int)(r / KTD);
    int n  = jt * 16 + (lane & 15);
    int k  = kt * 32 + ((i >> 3) << 4) + ((lane >> 4) & 1) * 8 + (i & 7);
    gstoreh(dst + f, (n < FRAME) ? f32_to_bf16(gloadf(Wdec + (size_t)n * HID + k))
                                 : (unsigned short)0);
  }
}

// ---------------- one LSTM layer phase (GEMM + gates, in-register) ----------------
// Software-pipelined with TWO statically-named fragment sets (no runtime-indexed
// arrays -> no scratch spills). KT is even for all call sites.
__device__ void layer_phase(int w, int lane, int KT,
                            const unsigned short* __restrict__ pb,
                            const float* __restrict__ bias,
                            float* __restrict__ cst,
                            const unsigned short* __restrict__ regX, int widthX, int ktX,
                            const unsigned short* __restrict__ regH,
                            unsigned short* __restrict__ dA,
                            unsigned short* __restrict__ dH) {
  v8f zero = {0.f,0.f,0.f,0.f,0.f,0.f,0.f,0.f};
  v8f acc[2][4];
#pragma unroll
  for (int mt = 0; mt < 2; ++mt)
#pragma unroll
    for (int g = 0; g < 4; ++g) acc[mt][g] = zero;

  // per-lane base pointers (A layout: lanes<16 K{0..7,16..23}, lanes>=16 +8)
  const int row  = lane & 15;
  const int koff = ((lane >> 4) & 1) * 8;
  const unsigned short* pX0 = regX + (size_t)row * widthX + koff;
  const unsigned short* pX1 = regX + (size_t)(row + 16) * widthX + koff;
  const unsigned short* pH0 = regH + (size_t)row * HID + koff;
  const unsigned short* pH1 = regH + (size_t)(row + 16) * HID + koff;
  const unsigned short* pbl = pb + (size_t)w * KT * 2048 + (size_t)lane * 16;

  FragSet fe, fo;

  auto LOAD = [&](int kt, FragSet& f) {
    const unsigned short *p0, *p1;
    if (kt < ktX) { p0 = pX0 + kt * 32;          p1 = pX1 + kt * 32; }
    else          { int kk = (kt - ktX) * 32;    p0 = pH0 + kk; p1 = pH1 + kk; }
    f.a0.q[0] = gload4(p0);  f.a0.q[1] = gload4(p0 + 16);
    f.a1.q[0] = gload4(p1);  f.a1.q[1] = gload4(p1 + 16);
    const unsigned short* pk = pbl + (size_t)kt * 2048;
    f.b0.q[0] = gload4(pk);        f.b0.q[1] = gload4(pk + 8);
    f.b1.q[0] = gload4(pk + 512);  f.b1.q[1] = gload4(pk + 520);
    f.b2.q[0] = gload4(pk + 1024); f.b2.q[1] = gload4(pk + 1032);
    f.b3.q[0] = gload4(pk + 1536); f.b3.q[1] = gload4(pk + 1544);
  };
  auto MM = [&](FragSet& f) {
    acc[0][0] = __builtin_amdgcn_wmma_f32_16x16x32_bf16(false, f.a0.v, false, f.b0.v, (short)0, acc[0][0], false, false);
    acc[1][0] = __builtin_amdgcn_wmma_f32_16x16x32_bf16(false, f.a1.v, false, f.b0.v, (short)0, acc[1][0], false, false);
    acc[0][1] = __builtin_amdgcn_wmma_f32_16x16x32_bf16(false, f.a0.v, false, f.b1.v, (short)0, acc[0][1], false, false);
    acc[1][1] = __builtin_amdgcn_wmma_f32_16x16x32_bf16(false, f.a1.v, false, f.b1.v, (short)0, acc[1][1], false, false);
    acc[0][2] = __builtin_amdgcn_wmma_f32_16x16x32_bf16(false, f.a0.v, false, f.b2.v, (short)0, acc[0][2], false, false);
    acc[1][2] = __builtin_amdgcn_wmma_f32_16x16x32_bf16(false, f.a1.v, false, f.b2.v, (short)0, acc[1][2], false, false);
    acc[0][3] = __builtin_amdgcn_wmma_f32_16x16x32_bf16(false, f.a0.v, false, f.b3.v, (short)0, acc[0][3], false, false);
    acc[1][3] = __builtin_amdgcn_wmma_f32_16x16x32_bf16(false, f.a1.v, false, f.b3.v, (short)0, acc[1][3], false, false);
  };

  LOAD(0, fe);
  for (int kt = 0; kt < KT; kt += 2) {       // KT even
    LOAD(kt + 1, fo);                        // prefetch odd
    MM(fe);
    LOAD((kt + 2 < KT) ? kt + 2 : 0, fe);    // prefetch next even (tail: dummy)
    MM(fo);
  }

  // elementwise LSTM cell: accumulator (r, lane) holds (M, N), identical layout
  // across the four gate accumulators -> fully in-register gate math.
  int nloc = (w << 4) + (lane & 15);
  float bi  = gloadf(bias + nloc);
  float bfv = gloadf(bias + HID + nloc);
  float bg  = gloadf(bias + 2 * HID + nloc);
  float bo  = gloadf(bias + 3 * HID + nloc);
  int mhi = (lane & 16) ? 8 : 0;
#pragma unroll
  for (int mt = 0; mt < 2; ++mt) {
#pragma unroll
    for (int r = 0; r < 8; ++r) {
      int m = (mt << 4) + mhi + r;
      size_t idx = (size_t)m * HID + nloc;
      float gi = acc[mt][0][r] + bi;
      float gf = acc[mt][1][r] + bfv;
      float gg = acc[mt][2][r] + bg;
      float go = acc[mt][3][r] + bo;
      float cn = sigm(gf) * gloadf(cst + idx) + sigm(gi) * tanhf(gg);
      float hn = sigm(go) * tanhf(cn);
      gstoref(cst + idx, cn);
      unsigned short hb = f32_to_bf16(hn);
      gstoreh(dA + idx, hb);   // next layer's A input
      gstoreh(dH + idx, hb);   // own recurrent input (next step, ping-pong)
    }
  }
}

// ---------------- persistent kernel: all 100 steps ----------------
__global__ void __launch_bounds__(NTHREADS, 1)
lstm_persistent(char* __restrict__ ws, float* __restrict__ dout, const int* __restrict__ genp) {
  const int lane = threadIdx.x & 31;
  const int w    = blockIdx.x * (NTHREADS / 32) + (threadIdx.x >> 5);
  const int gtid = blockIdx.x * NTHREADS + threadIdx.x;

  const unsigned short* pb1 = (const unsigned short*)(ws + OFF_PB1);
  const unsigned short* pb2 = (const unsigned short*)(ws + OFF_PB2);
  const unsigned short* pb3 = (const unsigned short*)(ws + OFF_PB3);
  const unsigned short* pbd = (const unsigned short*)(ws + OFF_PBD);
  const float* b1 = (const float*)(ws + OFF_B1);
  const float* b2 = (const float*)(ws + OFF_B2);
  const float* b3 = (const float*)(ws + OFF_B3);
  const float* bd = (const float*)(ws + OFF_BD);
  unsigned short* X1    = (unsigned short*)(ws + OFF_X1);
  unsigned short* H1    = (unsigned short*)(ws + OFF_H1);
  unsigned short* A2H1  = (unsigned short*)(ws + OFF_A2H1);
  unsigned short* H2    = (unsigned short*)(ws + OFF_H2);
  unsigned short* A3H2  = (unsigned short*)(ws + OFF_A3H2);
  unsigned short* H3    = (unsigned short*)(ws + OFF_H3);
  unsigned short* ADEC  = (unsigned short*)(ws + OFF_ADEC);
  const unsigned short* initx = (const unsigned short*)(ws + OFF_INITX);
  float* c1 = (float*)(ws + OFF_C);
  float* c2 = c1 + (size_t)BATCH * HID;
  float* c3 = c2 + (size_t)BATCH * HID;
  int* cnt = (int*)(ws + OFF_SYNC);
  int* phs = cnt + 1;

  const size_t HSZ = (size_t)BATCH * HID;
  const int steps  = T0 + genp[0];
  const size_t orow = (size_t)steps * FRAME;
  int target = 0;

  for (int t = 0; t < steps; ++t) {
    const int pr = t & 1, pw = pr ^ 1;

    // ---- layer 1: A = [x(t) | h1(t-1)] ----
    layer_phase(w, lane, KT1, pb1, b1, c1, X1, KX1, KX1/32, H1 + (size_t)pr*HSZ,
                A2H1, H1 + (size_t)pw*HSZ);
    ++target; grid_sync(cnt, phs, target);

    // ---- layer 2: A = [h1(t) | h2(t-1)] ----
    layer_phase(w, lane, KT2, pb2, b2, c2, A2H1, HID, HID/32, H2 + (size_t)pr*HSZ,
                A3H2, H2 + (size_t)pw*HSZ);
    ++target; grid_sync(cnt, phs, target);

    // ---- layer 3: A = [h2(t) | h3(t-1)] ----
    layer_phase(w, lane, KT2, pb3, b3, c3, A3H2, HID, HID/32, H3 + (size_t)pr*HSZ,
                ADEC, H3 + (size_t)pw*HSZ);
    ++target; grid_sync(cnt, phs, target);

    // ---- decoder: out = h3 @ Wdec^T + bdec (pipelined, even/odd sets) ----
    if (w < NTD) {
      v8f zero = {0.f,0.f,0.f,0.f,0.f,0.f,0.f,0.f};
      v8f a0c = zero, a1c = zero;
      const int row  = lane & 15;
      const int koff = ((lane >> 4) & 1) * 8;
      const unsigned short* pA0 = ADEC + (size_t)row * HID + koff;
      const unsigned short* pA1 = ADEC + (size_t)(row + 16) * HID + koff;
      const unsigned short* pbl = pbd + (size_t)w * KTD * 512 + (size_t)lane * 16;
      DFragSet fe, fo;
      auto DLOAD = [&](int kt, DFragSet& f) {
        const unsigned short* p0 = pA0 + kt * 32;
        const unsigned short* p1 = pA1 + kt * 32;
        f.a0.q[0] = gload4(p0);  f.a0.q[1] = gload4(p0 + 16);
        f.a1.q[0] = gload4(p1);  f.a1.q[1] = gload4(p1 + 16);
        const unsigned short* pk = pbl + (size_t)kt * 512;
        f.b.q[0] = gload4(pk);
        f.b.q[1] = gload4(pk + 8);
      };
      auto DMM = [&](DFragSet& f) {
        a0c = __builtin_amdgcn_wmma_f32_16x16x32_bf16(false, f.a0.v, false, f.b.v, (short)0, a0c, false, false);
        a1c = __builtin_amdgcn_wmma_f32_16x16x32_bf16(false, f.a1.v, false, f.b.v, (short)0, a1c, false, false);
      };
      DLOAD(0, fe);
      for (int kt = 0; kt < KTD; kt += 2) {      // KTD even
        DLOAD(kt + 1, fo);
        DMM(fe);
        DLOAD((kt + 2 < KTD) ? kt + 2 : 0, fe);
        DMM(fo);
      }
      int n = (w << 4) + (lane & 15);
      float bb = gloadf(bd + n);
      int mhi = (lane & 16) ? 8 : 0;
#pragma unroll
      for (int mt = 0; mt < 2; ++mt) {
#pragma unroll
        for (int r = 0; r < 8; ++r) {
          int m = (mt << 4) + mhi + r;
          float v = (mt ? a1c[r] : a0c[r]) + bb;
          if (n < FRAME) {
            gstoref(dout + (size_t)m * orow + (size_t)t * FRAME + n, v);
            if (t >= T0 - 1)   // autoregressive feedback for step t+1
              gstoreh(X1 + (size_t)m * KX1 + n, f32_to_bf16(v));
          }
        }
      }
    }
    // priming: next x comes from initial_seq
    if (t + 1 < T0) {
      for (int i = gtid; i < BATCH * KX1; i += NBLOCKS * NTHREADS)
        gstoreh(X1 + i, gloadh(initx + (size_t)(t + 1) * BATCH * KX1 + i));
    }
    ++target; grid_sync(cnt, phs, target);
  }
}

// ---------------- launcher ----------------
extern "C" void kernel_launch(void* const* d_in, const int* in_sizes, int n_in,
                              void* d_out, int out_size, void* d_ws, size_t ws_size,
                              hipStream_t stream) {
  (void)in_sizes; (void)n_in; (void)out_size;
  if (ws_size < WS_NEED) return;   // need ~45 MB scratch

  const float* iseq = (const float*)d_in[0];
  const float* Wih1 = (const float*)d_in[1];
  const float* Whh1 = (const float*)d_in[2];
  const float* bih1 = (const float*)d_in[3];
  const float* bhh1 = (const float*)d_in[4];
  const float* Wih2 = (const float*)d_in[5];
  const float* Whh2 = (const float*)d_in[6];
  const float* bih2 = (const float*)d_in[7];
  const float* bhh2 = (const float*)d_in[8];
  const float* Wih3 = (const float*)d_in[9];
  const float* Whh3 = (const float*)d_in[10];
  const float* bih3 = (const float*)d_in[11];
  const float* bhh3 = (const float*)d_in[12];
  const float* Wdec = (const float*)d_in[13];
  const float* bdec = (const float*)d_in[14];
  const int*   gen  = (const int*)d_in[15];

  char* ws = (char*)d_ws;
  Ptrs p{iseq, bih1, bhh1, bih2, bhh2, bih3, bhh3, bdec};

  misc_init<<<256, 256, 0, stream>>>(p, ws);
  pack_gates<<<2048, 256, 0, stream>>>(Wih1, FRAME, KX1, FRAME, Whh1, KT1,
                                       (unsigned short*)(ws + OFF_PB1), PB1_E);
  pack_gates<<<2048, 256, 0, stream>>>(Wih2, HID, HID, HID, Whh2, KT2,
                                       (unsigned short*)(ws + OFF_PB2), PB2_E);
  pack_gates<<<2048, 256, 0, stream>>>(Wih3, HID, HID, HID, Whh3, KT2,
                                       (unsigned short*)(ws + OFF_PB3), PB2_E);
  pack_dec<<<512, 256, 0, stream>>>(Wdec, (unsigned short*)(ws + OFF_PBD));

  lstm_persistent<<<NBLOCKS, NTHREADS, 0, stream>>>(ws, (float*)d_out, gen);
}
